// EmlInspiredTreeFeedForward_47064251630325
// MI455X (gfx1250) — compile-verified, matrix operand order checked by reading, not run on hardware
//
#include <hip/hip_runtime.h>

// ---------- types ----------
typedef _Float16 v16h __attribute__((ext_vector_type(16)));
typedef _Float16 v8h  __attribute__((ext_vector_type(8)));
typedef _Float16 v4h  __attribute__((ext_vector_type(4)));
typedef float    v8f  __attribute__((ext_vector_type(8)));
typedef float    v4f  __attribute__((ext_vector_type(4)));
typedef int      v4i  __attribute__((ext_vector_type(4)));

#define AS_GLOBAL __attribute__((address_space(1)))
#define AS_LOCAL  __attribute__((address_space(3)))

// ---------- problem dims ----------
#define MDIM 4096   // B*S
#define KDIM 4096   // W
#define NDIM 1024   // D
#define WNUM 4096
#define DDIM 1024

// ---------- async-to-LDS (probe-verified on this toolchain) ----------
#if __has_builtin(__builtin_amdgcn_global_load_async_to_lds_b128)
#define HAVE_ASYNC 1
#endif

__device__ __forceinline__ void async_b128(const _Float16* gp, _Float16* lp) {
#ifdef HAVE_ASYNC
  __builtin_amdgcn_global_load_async_to_lds_b128((AS_GLOBAL v4i*)gp,
                                                 (AS_LOCAL v4i*)lp, 0, 0);
#else
  *(v8h*)lp = *(const v8h*)gp;
#endif
}

__device__ __forceinline__ void wait_async_zero() {
#ifdef HAVE_ASYNC
#if __has_builtin(__builtin_amdgcn_s_wait_asynccnt)
  __builtin_amdgcn_s_wait_asynccnt(0);
#else
  asm volatile("s_wait_asynccnt 0" ::: "memory");
#endif
#endif
}

// ============================================================
// K1: selector = softmax(leaf_logits) folded against the const
//     basis tail (-1,0,1):  selW[w][l][0..7], selB[w][l]
// ============================================================
__global__ __launch_bounds__(256) void k_selector(const float* __restrict__ logits,
                                                  float* __restrict__ selW,
                                                  float* __restrict__ selB) {
  int idx = blockIdx.x * 256 + threadIdx.x;      // (w*8 + l), 32768 rows
  if (idx >= WNUM * 8) return;
  const float* p = logits + (size_t)idx * 11;
  float x[11], mx = -3.4e38f;
#pragma unroll
  for (int c = 0; c < 11; ++c) { x[c] = p[c]; mx = fmaxf(mx, x[c]); }
  float s = 0.f;
#pragma unroll
  for (int c = 0; c < 11; ++c) { x[c] = __expf(x[c] - mx); s += x[c]; }
  float inv = 1.f / s;
#pragma unroll
  for (int c = 0; c < 8; ++c) selW[(size_t)idx * 8 + c] = x[c] * inv;
  selB[idx] = (x[10] - x[8]) * inv;              // (+1)*sel[10] + (-1)*sel[8]
}

// ============================================================
// K2: basis[m][j] = tanh(hidden[m,:] . slot_w[:,j] + slot_b[j])
//     one wave per slot (8 waves / block, one block per m)
// ============================================================
__global__ __launch_bounds__(256) void k_basis(const float* __restrict__ hidden,
                                               const float* __restrict__ slot_w,
                                               const float* __restrict__ slot_b,
                                               float* __restrict__ basis) {
  const int m = blockIdx.x;
  const int j = threadIdx.x >> 5;                // slot 0..7
  const int lane = threadIdx.x & 31;
  const float* h = hidden + (size_t)m * DDIM;
  float acc = 0.f;
  for (int d = lane; d < DDIM; d += 32)
    acc += h[d] * slot_w[d * 8 + j];
#pragma unroll
  for (int off = 16; off > 0; off >>= 1)
    acc += __shfl_down(acc, off, 32);
  if (lane == 0) basis[m * 8 + j] = tanhf(acc + slot_b[j]);
}

// ============================================================
// K3: out_w (f32, K x N row-major) -> whT (f16, N x K row-major)
//     LDS-tiled 32x32 transpose so both global accesses coalesce.
// ============================================================
__global__ __launch_bounds__(256) void k_convT(const float* __restrict__ w,
                                               _Float16* __restrict__ whT) {
  __shared__ _Float16 tile[32][33];              // [n][k], padded
  const int t  = threadIdx.x;
  const int kb = blockIdx.x * 32;
  const int nb = blockIdx.y * 32;
  const int tx = t & 31, ty = t >> 5;            // ty = 0..7
#pragma unroll
  for (int i = ty; i < 32; i += 8)               // read rows k=kb+i, coalesced in n
    tile[tx][i] = (_Float16)w[(size_t)(kb + i) * NDIM + nb + tx];
  __syncthreads();
#pragma unroll
  for (int i = ty; i < 32; i += 8)               // write rows n=nb+i, coalesced in k
    whT[(size_t)(nb + i) * KDIM + kb + tx] = tile[i][tx];
}

// ============================================================
// K4: fused leaf-values + depth-3 tree -> roots (f16, M x K)
//     tile: 128 w  x 32 m per block; selector tile staged in LDS
//     and reused by all 32 m-rows.
// ============================================================
__global__ __launch_bounds__(256) void k_roots(const float* __restrict__ selW,
                                               const float* __restrict__ selB,
                                               const float* __restrict__ basis,
                                               const float* __restrict__ np,
                                               _Float16* __restrict__ roots) {
  __shared__ float sW[128 * 64];
  __shared__ float sB[128 * 8];
  __shared__ float sBa[32 * 8];
  const int t  = threadIdx.x;
  const int w0 = blockIdx.x * 128;
  const int m0 = blockIdx.y * 32;
  for (int i = t; i < 128 * 64; i += 256) sW[i]  = selW[(size_t)w0 * 64 + i];
  for (int i = t; i < 128 * 8;  i += 256) sB[i]  = selB[(size_t)w0 * 8 + i];
  for (int i = t; i < 32 * 8;   i += 256) sBa[i] = basis[(size_t)m0 * 8 + i];
  const float lw = np[0], rw = np[1], pw = np[2], dw = np[3], nb = np[4];
  __syncthreads();

  const int wi = t & 127;
  for (int mi = t >> 7; mi < 32; mi += 2) {
    float bs[8];
#pragma unroll
    for (int c = 0; c < 8; ++c) bs[c] = sBa[mi * 8 + c];
    float v[8];
#pragma unroll
    for (int l = 0; l < 8; ++l) {
      float a = sB[wi * 8 + l];
      const float* sp = &sW[(wi * 8 + l) * 8];
#pragma unroll
      for (int c = 0; c < 8; ++c) a += sp[c] * bs[c];
      v[l] = a;
    }
#pragma unroll
    for (int dpt = 0; dpt < 3; ++dpt) {
      const int n = 4 >> dpt;
#pragma unroll
      for (int i = 0; i < n; ++i) {
        float L = v[2 * i], R = v[2 * i + 1];
        v[i] = tanhf(lw * L + rw * R + pw * L * R + dw * (L - R) + nb);
      }
    }
    roots[(size_t)(m0 + mi) * (size_t)KDIM + w0 + wi] = (_Float16)v[0];
  }
}

// ============================================================
// K5: out = roots(f16, MxK) @ out_w(f16) + out_b     (f32 out)
//     B is pre-transposed (whT, NxK) so BOTH tiles stage with
//     global_load_async_to_lds_b128 (contiguous along K).
//     Block tile 128x128, 8 waves as 4(M) x 2(N); each wave owns
//     32x64 = 2x4 WMMA fragments -> 8 v_wmma per K-step of 32.
//     LDS row stride 40 halves = 80B keeps b128 16B alignment.
// ============================================================
__global__ __launch_bounds__(256) void k_gemm(const _Float16* __restrict__ A,
                                              const _Float16* __restrict__ BT,
                                              const float* __restrict__ bias,
                                              float* __restrict__ Cout) {
  __shared__ _Float16 As[2][128 * 40];   // [m][k]
  __shared__ _Float16 Bs[2][128 * 40];   // [n][k]
  const int t    = threadIdx.x;
  const int m0   = blockIdx.x * 128;
  const int n0   = blockIdx.y * 128;
  const int wave = t >> 5, lane = t & 31;
  const int wm   = wave & 3;          // 4 bands of 32 rows (M)
  const int wn   = wave >> 2;         // 2 bands of 64 cols (N)
  const int h    = lane >> 4;         // half-wave selects K-octet / M+8
  const int rr   = lane & 15;         // M for A-frag, N for B-frag

  v8f acc[2][4] = {};

  auto stage = [&](int buf, int kt) {
    const int k0 = kt * 32;
    // each tile: 128 rows x 32 halves = 512 b128 chunks, 2 per thread
#pragma unroll
    for (int it = 0; it < 2; ++it) {
      const int id = t + it * 256;
      const int r = id >> 2, c8 = (id & 3) * 8;
      async_b128(&A [(size_t)(m0 + r) * KDIM + k0 + c8], &As[buf][r * 40 + c8]);
      async_b128(&BT[(size_t)(n0 + r) * KDIM + k0 + c8], &Bs[buf][r * 40 + c8]);
    }
  };

  stage(0, 0);
  wait_async_zero();
  __syncthreads();

  for (int kt = 0; kt < KDIM / 32; ++kt) {
    const int cur = kt & 1;
    if (kt + 1 < KDIM / 32) stage(cur ^ 1, kt + 1);
    if (kt + 2 < KDIM / 32) {  // global_prefetch_b8 on gfx1250
      __builtin_prefetch(&A [(size_t)(m0 + (t >> 1)) * KDIM + (size_t)(kt + 2) * 32], 0, 1);
      __builtin_prefetch(&BT[(size_t)(n0 + (t >> 1)) * KDIM + (size_t)(kt + 2) * 32], 0, 1);
    }

    // fragment loads: element e<8 -> K = 8h+e ; e>=8 -> K = 16+8h+(e-8)
    v16h af[2], bf[4];
#pragma unroll
    for (int fm = 0; fm < 2; ++fm) {
      const int row = (wm * 32 + fm * 16 + rr) * 40;
      v8h lo = *(const v8h*)&As[cur][row + 8 * h];
      v8h hi = *(const v8h*)&As[cur][row + 16 + 8 * h];
      af[fm] = __builtin_shufflevector(lo, hi, 0,1,2,3,4,5,6,7,8,9,10,11,12,13,14,15);
    }
#pragma unroll
    for (int fn = 0; fn < 4; ++fn) {
      const int row = (wn * 64 + fn * 16 + rr) * 40;
      v8h lo = *(const v8h*)&Bs[cur][row + 8 * h];
      v8h hi = *(const v8h*)&Bs[cur][row + 16 + 8 * h];
      bf[fn] = __builtin_shufflevector(lo, hi, 0,1,2,3,4,5,6,7,8,9,10,11,12,13,14,15);
    }
#pragma unroll
    for (int fm = 0; fm < 2; ++fm)
#pragma unroll
      for (int fn = 0; fn < 4; ++fn)
        acc[fm][fn] = __builtin_amdgcn_wmma_f32_16x16x32_f16(
            false, af[fm], false, bf[fn], (short)0, acc[fm][fn], false, false);

    wait_async_zero();
    __syncthreads();
  }

  // epilogue: C/D layout -> lane rr = N, VGPR r + 8h = M
#pragma unroll
  for (int fm = 0; fm < 2; ++fm) {
#pragma unroll
    for (int fn = 0; fn < 4; ++fn) {
      const int col = n0 + wn * 64 + fn * 16 + rr;
      const float bv = bias[col];
#pragma unroll
      for (int r = 0; r < 8; ++r) {
        const int row = m0 + wm * 32 + fm * 16 + h * 8 + r;
        Cout[(size_t)row * NDIM + col] = acc[fm][fn][r] + bv;
      }
    }
  }
}

// ============================================================
// launch
// ============================================================
extern "C" void kernel_launch(void* const* d_in, const int* in_sizes, int n_in,
                              void* d_out, int out_size, void* d_ws, size_t ws_size,
                              hipStream_t stream) {
  (void)in_sizes; (void)n_in; (void)out_size; (void)ws_size;
  const float* hidden      = (const float*)d_in[0];
  const float* slot_w      = (const float*)d_in[1];
  const float* slot_b      = (const float*)d_in[2];
  const float* leaf_logits = (const float*)d_in[3];
  const float* node_params = (const float*)d_in[4];
  const float* out_w       = (const float*)d_in[5];
  const float* out_b       = (const float*)d_in[6];
  float* out = (float*)d_out;

  // workspace layout (bytes), all offsets 256B-aligned
  char* ws = (char*)d_ws;
  float*    selW  = (float*)(ws + 0);              // 4096*64*4   = 1,048,576
  float*    selB  = (float*)(ws + 1048576);        // 4096*8*4    =   131,072
  float*    basis = (float*)(ws + 1179648);        // 4096*8*4    =   131,072
  _Float16* roots = (_Float16*)(ws + 1310720);     // 4096*4096*2 = 33,554,432
  _Float16* outwT = (_Float16*)(ws + 34865152);    // 1024*4096*2 =  8,388,608

  hipLaunchKernelGGL(k_selector, dim3(128),      dim3(256), 0, stream,
                     leaf_logits, selW, selB);
  hipLaunchKernelGGL(k_basis,    dim3(4096),     dim3(256), 0, stream,
                     hidden, slot_w, slot_b, basis);
  hipLaunchKernelGGL(k_convT,    dim3(128, 32),  dim3(256), 0, stream,
                     out_w, outwT);
  hipLaunchKernelGGL(k_roots,    dim3(32, 128),  dim3(256), 0, stream,
                     selW, selB, basis, node_params, roots);
  hipLaunchKernelGGL(k_gemm,     dim3(32, 8),    dim3(256), 0, stream,
                     roots, outwT, out_b, out);
}
